// GPT_42752104464534
// MI455X (gfx1250) — matrix operand order repaired and checked by reading, not compiled, not run on hardware
//
#include <hip/hip_runtime.h>
#include <hip/hip_bf16.h>

// ---------------------------------------------------------------------------
// GPT forward for MI455X (gfx1250, wave32, WMMA).
// bf16 weights packed once into WMMA-B fragment layout (fits 192MB L2),
// all matmuls via v_wmma_f32_16x16x32_bf16 with 32x32 per-wave tiles,
// flash-style fused attention, logits+softmax straight into d_out.
// ---------------------------------------------------------------------------

typedef __attribute__((ext_vector_type(16))) __bf16 v16bf;
typedef __attribute__((ext_vector_type(8)))  __bf16 v8bf;
typedef __attribute__((ext_vector_type(8)))  float  v8f;

namespace {
constexpr int kV = 32000;
constexpr int kD = 1024;
constexpr int kL = 12;
constexpr int kH = 16;
constexpr int kHD = 64;
constexpr int kB = 2;
constexpr int kT = 1024;
constexpr int kM = kB * kT;          // 2048 token rows
} // namespace

// ---------------------------------------------------------------------------
// Fragment loaders (contiguous k order k = 16*(lane/16) + s for A and B, wave32)
// ---------------------------------------------------------------------------

// 16-bit A fragment 16x32 (MxK), from row-major storage.
// lane: m = lane&15; khw = lane<16 ? 0 : 16.
// halves 0..7  -> K = kbase+khw+0..7   (one b128)
// halves 8..15 -> K = kbase+khw+8..15  (one b128)
__device__ __forceinline__ v16bf load_fragA(const __bf16* base, int row0,
                                            int rstride, int kbase) {
  const int lane = threadIdx.x & 31;
  const __bf16* p = base + (size_t)(row0 + (lane & 15)) * rstride + kbase +
                    ((lane < 16) ? 0 : 16);
  v8bf lo = *(const v8bf*)p;
  v8bf hi = *(const v8bf*)(p + 8);
  v16bf r;
#pragma unroll
  for (int i = 0; i < 8; ++i) { r[i] = lo[i]; r[i + 8] = hi[i]; }
  return r;
}

// 16-bit B fragment 32x16 (KxN) where the K dimension is contiguous in memory
// (i.e. we hold B^T row-major: mat[n][k]).  lane: n = col0 + (lane&15);
// halves 0..15 -> K = kbase + (lane<16?0:16) + 0..15 (two b128, contiguous).
__device__ __forceinline__ v16bf load_fragB_rm(const __bf16* base, int col0,
                                               int rstride, int kbase) {
  const int lane = threadIdx.x & 31;
  const __bf16* p = base + (size_t)(col0 + (lane & 15)) * rstride + kbase +
                    ((lane < 16) ? 0 : 16);
  v8bf lo = *(const v8bf*)p;
  v8bf hi = *(const v8bf*)(p + 8);
  v16bf r;
#pragma unroll
  for (int i = 0; i < 8; ++i) { r[i] = lo[i]; r[i + 8] = hi[i]; }
  return r;
}

// B fragment from the pre-packed weight image: 1024B per 32x16 tile,
// chunk0 (halves 0..7) at lane*16B, chunk1 (halves 8..15) at 512B + lane*16B.
// Fully coalesced 512B per wave per load.
__device__ __forceinline__ v16bf load_fragPK(const __bf16* tile) {
  const int lane = threadIdx.x & 31;
  const __bf16* p = tile + lane * 8;
  v8bf lo = *(const v8bf*)p;
  v8bf hi = *(const v8bf*)(p + 256);
  v16bf r;
#pragma unroll
  for (int i = 0; i < 8; ++i) { r[i] = lo[i]; r[i + 8] = hi[i]; }
  return r;
}

__device__ __forceinline__ v8f wmma_bf16(v16bf a, v16bf b, v8f c) {
  return __builtin_amdgcn_wmma_f32_16x16x32_bf16(
      /*neg_a=*/false, a, /*neg_b=*/false, b,
      /*c_mod=*/(short)0, c, /*reuse_a=*/false, /*reuse_b=*/false);
}

// ---------------------------------------------------------------------------
// Weight repack: f32 [nmat, K, N] row-major -> bf16 WMMA-B fragment image.
// Tile order: tile = nt*(K/32) + kt  (kt contiguous for streaming).
// ---------------------------------------------------------------------------
__global__ void __launch_bounds__(256)
pack_w_kernel(const float* __restrict__ src, __bf16* __restrict__ dst,
              int Kd, int Nd, int nmat) {
  size_t gid = (size_t)blockIdx.x * 256 + threadIdx.x;
  size_t per = (size_t)Kd * Nd;
  if (gid >= per * (size_t)nmat) return;
  int mat = (int)(gid / per);
  size_t rem = gid - (size_t)mat * per;
  int tile = (int)(rem >> 9);
  int w = (int)(rem & 511);
  int Kt = Kd >> 5;
  int nt = tile / Kt;
  int kt = tile - nt * Kt;
  int c = w >> 8;          // chunk (halves 0..7 / 8..15)
  int r8 = w & 255;
  int lane = r8 >> 3;
  int i = r8 & 7;
  int kk = kt * 32 + ((lane < 16) ? 0 : 16) + c * 8 + i;
  int nn = nt * 16 + (lane & 15);
  dst[gid] = (__bf16)src[(size_t)mat * per + (size_t)kk * Nd + nn];
}

// ---------------------------------------------------------------------------
// Embedding: x[b,t,:] = tok_emb[ctx[b,t],:] + pos_emb[t,:]
// ---------------------------------------------------------------------------
__global__ void __launch_bounds__(256)
embed_kernel(const int* __restrict__ ctx, const float* __restrict__ tok,
             const float* __restrict__ pos, float* __restrict__ x) {
  int row = blockIdx.x;                 // 0..2047
  int t = row & (kT - 1);
  int id = ctx[row];
  const float* te = tok + (size_t)id * kD;
  const float* pe = pos + (size_t)t * kD;
  float* xr = x + (size_t)row * kD;
  for (int i = threadIdx.x; i < kD; i += 256) xr[i] = te[i] + pe[i];
}

// ---------------------------------------------------------------------------
// LayerNorm (f32 in, bf16 out), one block per row of D=1024.
// ---------------------------------------------------------------------------
__global__ void __launch_bounds__(256)
ln_kernel(const float* __restrict__ xin, const float* __restrict__ sc,
          const float* __restrict__ bi, __bf16* __restrict__ out) {
  int row = blockIdx.x;
  const float* xr = xin + (size_t)row * kD;
  int tid = threadIdx.x, lane = tid & 31, wid = tid >> 5;
  float s = 0.f, s2 = 0.f;
  for (int i = tid; i < kD; i += 256) { float v = xr[i]; s += v; s2 += v * v; }
#pragma unroll
  for (int d = 1; d < 32; d <<= 1) {
    s += __shfl_xor(s, d, 32);
    s2 += __shfl_xor(s2, d, 32);
  }
  __shared__ float rs[8], rs2[8], stats[2];
  if (lane == 0) { rs[wid] = s; rs2[wid] = s2; }
  __syncthreads();
  if (tid == 0) {
    float ts = 0.f, ts2 = 0.f;
#pragma unroll
    for (int i = 0; i < 8; ++i) { ts += rs[i]; ts2 += rs2[i]; }
    float mean = ts / (float)kD;
    float var = ts2 / (float)kD - mean * mean;
    stats[0] = mean;
    stats[1] = rsqrtf(var + 1e-5f);
  }
  __syncthreads();
  float mean = stats[0], rstd = stats[1];
  __bf16* orow = out + (size_t)row * kD;
  for (int i = tid; i < kD; i += 256)
    orow[i] = (__bf16)((xr[i] - mean) * rstd * sc[i] + bi[i]);
}

// ---------------------------------------------------------------------------
// GEMM epilogue for one 16x16 C tile.
// OM: 0 = f32 row-major, 1 = bf16 row-major, 2 = bf16 V-transposed [B,H,HD,T]
// ---------------------------------------------------------------------------
template <int OM, bool BIAS, bool RELU, bool RES>
__device__ __forceinline__ void store_tile(v8f acc, int mt, int nt,
                                           const float* __restrict__ bias,
                                           const float* __restrict__ res,
                                           void* __restrict__ outp, int Nc) {
  const int lane = threadIdx.x & 31;
  int n = nt * 16 + (lane & 15);
  int mbase = mt * 16 + ((lane < 16) ? 0 : 8);
  float bv = BIAS ? bias[n] : 0.0f;
#pragma unroll
  for (int j = 0; j < 8; ++j) {
    int m = mbase + j;
    float v = acc[j] + bv;
    if (RELU) v = fmaxf(v, 0.0f);
    if (RES) v += res[(size_t)m * Nc + n];
    if (OM == 0) {
      ((float*)outp)[(size_t)m * Nc + n] = v;
    } else if (OM == 1) {
      ((__bf16*)outp)[(size_t)m * Nc + n] = (__bf16)v;
    } else {
      // V stored transposed: [B, H, HD, T]  (m = b*T+t over 2048, n = h*64+hd)
      int bb = m >> 10, tt = m & 1023;
      int hh = n >> 6, hd = n & 63;
      ((__bf16*)outp)[(((size_t)(bb * kH + hh) * kHD + hd) << 10) + tt] =
          (__bf16)v;
    }
  }
}

// ---------------------------------------------------------------------------
// WMMA GEMM: out[M,N] = A[M,K](bf16 rm) x Wp(packed bf16) (+bias)(+relu)(+res)
// 32x32 tile per wave (2 A-frags x 2 B-frags -> 4 WMMA per K-step).
// Block: 8 waves as 2(m) x 4(n) -> 64 x 128 block tile.
// grid = (Nc/128, Mr/64).
// ---------------------------------------------------------------------------
template <int OM, bool BIAS, bool RELU, bool RES>
__global__ void __launch_bounds__(256)
gemm_bf16(const __bf16* __restrict__ A, const __bf16* __restrict__ Wp,
          const float* __restrict__ bias, const float* __restrict__ res,
          void* __restrict__ outp, int Mr, int Nc, int Kd) {
  int wid = threadIdx.x >> 5;
  int wn = wid & 3;                    // 4 waves along n
  int wm = wid >> 2;                   // 2 waves along m
  int nt0 = blockIdx.x * 8 + wn * 2;   // units of 16 cols
  int mt0 = blockIdx.y * 4 + wm * 2;   // units of 16 rows
  if (nt0 * 16 >= Nc || mt0 * 16 >= Mr) return;
  int Kt = Kd >> 5;
  const __bf16* wt0 = Wp + (size_t)nt0 * Kt * 512;
  const __bf16* wt1 = Wp + (size_t)(nt0 + 1) * Kt * 512;
  v8f a00 = 0.0f, a01 = 0.0f, a10 = 0.0f, a11 = 0.0f;
  for (int kt = 0; kt < Kt; ++kt) {
    // prefetch packed-weight tiles a few K-steps ahead (global_prefetch path)
    int ktp = (kt + 4 < Kt) ? (kt + 4) : kt;
    __builtin_prefetch(wt0 + (size_t)ktp * 512, 0, 1);
    __builtin_prefetch(wt1 + (size_t)ktp * 512, 0, 1);
    v16bf fa0 = load_fragA(A, mt0 * 16, Kd, kt * 32);
    v16bf fa1 = load_fragA(A, mt0 * 16 + 16, Kd, kt * 32);
    v16bf fb0 = load_fragPK(wt0 + (size_t)kt * 512);
    v16bf fb1 = load_fragPK(wt1 + (size_t)kt * 512);
    a00 = wmma_bf16(fa0, fb0, a00);
    a01 = wmma_bf16(fa0, fb1, a01);
    a10 = wmma_bf16(fa1, fb0, a10);
    a11 = wmma_bf16(fa1, fb1, a11);
  }
  store_tile<OM, BIAS, RELU, RES>(a00, mt0, nt0, bias, res, outp, Nc);
  store_tile<OM, BIAS, RELU, RES>(a01, mt0, nt0 + 1, bias, res, outp, Nc);
  store_tile<OM, BIAS, RELU, RES>(a10, mt0 + 1, nt0, bias, res, outp, Nc);
  store_tile<OM, BIAS, RELU, RES>(a11, mt0 + 1, nt0 + 1, bias, res, outp, Nc);
}

// ---------------------------------------------------------------------------
// Fused causal attention (flash-style), one wave per 16-row query tile.
// q,k: bf16 [B*T, D] row-major; vT: bf16 [B,H,HD,T]; x: f32 residual (in/out).
// ---------------------------------------------------------------------------
__global__ void __launch_bounds__(128)
attn_kernel(const __bf16* __restrict__ q, const __bf16* __restrict__ k,
            const __bf16* __restrict__ vT, float* __restrict__ x) {
  int wid = threadIdx.x >> 5;
  int lane = threadIdx.x & 31;
  int mt = blockIdx.x * 4 + wid;        // 0..63 : query tile
  int h = blockIdx.y;
  int b = blockIdx.z;
  __shared__ __bf16 pls[4][16 * 32];
  __bf16* lp = pls[wid];

  const __bf16* qb = q + ((size_t)b * kT) * kD + h * kHD;
  const __bf16* kb = k + ((size_t)b * kT) * kD + h * kHD;
  const __bf16* vb = vT + ((size_t)(b * kH + h)) * kHD * kT;

  v16bf qa0 = load_fragA(qb, mt * 16, kD, 0);
  v16bf qa1 = load_fragA(qb, mt * 16, kD, 32);

  v8f o0 = 0.0f, o1 = 0.0f, o2 = 0.0f, o3 = 0.0f;
  float rm[8], rl[8];
#pragma unroll
  for (int j = 0; j < 8; ++j) { rm[j] = -1e30f; rl[j] = 0.0f; }

  const int nl = lane & 15;
  const int mo = (lane < 16) ? 0 : 8;
  const float scale = 0.125f;           // 1/sqrt(64)
  const int nkt2 = (mt * 16 + 15) / 32 + 1;

  for (int kt2 = 0; kt2 < nkt2; ++kt2) {
    const int ktA = kt2 * 2, ktB = ktA + 1;
    v8f sA = 0.0f, sB = 0.0f;
    sA = wmma_bf16(qa0, load_fragB_rm(kb, ktA * 16, kD, 0), sA);
    sA = wmma_bf16(qa1, load_fragB_rm(kb, ktA * 16, kD, 32), sA);
    sB = wmma_bf16(qa0, load_fragB_rm(kb, ktB * 16, kD, 0), sB);
    sB = wmma_bf16(qa1, load_fragB_rm(kb, ktB * 16, kD, 32), sB);

#pragma unroll
    for (int j = 0; j < 8; ++j) {
      int m = mt * 16 + j + mo;
      float a = sA[j] * scale;
      float c = sB[j] * scale;
      if (ktA * 16 + nl > m) a = -3.0e38f;    // causal mask
      if (ktB * 16 + nl > m) c = -3.0e38f;
      float amx = fmaxf(a, c);
#pragma unroll
      for (int d = 1; d < 16; d <<= 1) amx = fmaxf(amx, __shfl_xor(amx, d, 32));
      float mnew = fmaxf(rm[j], amx);
      float sc = __expf(rm[j] - mnew);
      float pa = __expf(a - mnew);
      float pc = __expf(c - mnew);
      float ps = pa + pc;
#pragma unroll
      for (int d = 1; d < 16; d <<= 1) ps += __shfl_xor(ps, d, 32);
      rl[j] = rl[j] * sc + ps;
      rm[j] = mnew;
      o0[j] *= sc; o1[j] *= sc; o2[j] *= sc; o3[j] *= sc;
      int rrow = j + mo;                       // C-layout -> row-major LDS tile
      lp[rrow * 32 + nl] = (__bf16)pa;
      lp[rrow * 32 + 16 + nl] = (__bf16)pc;
    }
    asm volatile("s_wait_dscnt 0" ::: "memory");
    v16bf pf = load_fragA(lp, 0, 32, 0);       // 16x32 probs as A fragment
    o0 = wmma_bf16(pf, load_fragB_rm(vb, 0,  kT, kt2 * 32), o0);
    o1 = wmma_bf16(pf, load_fragB_rm(vb, 16, kT, kt2 * 32), o1);
    o2 = wmma_bf16(pf, load_fragB_rm(vb, 32, kT, kt2 * 32), o2);
    o3 = wmma_bf16(pf, load_fragB_rm(vb, 48, kT, kt2 * 32), o3);
  }

#pragma unroll
  for (int j = 0; j < 8; ++j) {
    float inv = 1.0f / rl[j];
    int m = mt * 16 + j + mo;
    float* xr = x + ((size_t)b * kT + m) * kD + h * kHD + nl;
    xr[0]  += o0[j] * inv;                     // fused residual add
    xr[16] += o1[j] * inv;
    xr[32] += o2[j] * inv;
    xr[48] += o3[j] * inv;
  }
}

// ---------------------------------------------------------------------------
// Row softmax in place over V=32000, one block per row.
// ---------------------------------------------------------------------------
__global__ void __launch_bounds__(256)
softmax_kernel(float* __restrict__ out, int n) {
  int row = blockIdx.x;
  float* p = out + (size_t)row * n;
  int tid = threadIdx.x, lane = tid & 31, wid = tid >> 5;
  __shared__ float red[8];
  __shared__ float stat;
  float m = -3.0e38f;
  for (int i = tid; i < n; i += 256) m = fmaxf(m, p[i]);
#pragma unroll
  for (int d = 1; d < 32; d <<= 1) m = fmaxf(m, __shfl_xor(m, d, 32));
  if (lane == 0) red[wid] = m;
  __syncthreads();
  if (tid == 0) {
    float t = red[0];
#pragma unroll
    for (int i = 1; i < 8; ++i) t = fmaxf(t, red[i]);
    stat = t;
  }
  __syncthreads();
  m = stat;
  float s = 0.f;
  for (int i = tid; i < n; i += 256) {
    float e = __expf(p[i] - m);
    p[i] = e;
    s += e;
  }
#pragma unroll
  for (int d = 1; d < 32; d <<= 1) s += __shfl_xor(s, d, 32);
  __syncthreads();
  if (lane == 0) red[wid] = s;
  __syncthreads();
  if (tid == 0) {
    float t = 0.f;
#pragma unroll
    for (int i = 0; i < 8; ++i) t += red[i];
    stat = t;
  }
  __syncthreads();
  float inv = 1.0f / stat;
  for (int i = tid; i < n; i += 256) p[i] *= inv;
}

// ---------------------------------------------------------------------------
// Host-side orchestration
// ---------------------------------------------------------------------------
extern "C" void kernel_launch(void* const* d_in, const int* in_sizes, int n_in,
                              void* d_out, int out_size, void* d_ws,
                              size_t ws_size, hipStream_t stream) {
  (void)in_sizes; (void)n_in; (void)out_size; (void)ws_size;

  const int*   ctx   = (const int*)d_in[0];
  const float* tok   = (const float*)d_in[1];
  const float* pos   = (const float*)d_in[2];
  const float* wq    = (const float*)d_in[3];
  const float* wk    = (const float*)d_in[4];
  const float* wv    = (const float*)d_in[5];
  const float* ln1_s = (const float*)d_in[6];
  const float* ln1_b = (const float*)d_in[7];
  const float* ln2_s = (const float*)d_in[8];
  const float* ln2_b = (const float*)d_in[9];
  const float* fw1   = (const float*)d_in[10];
  const float* fb1   = (const float*)d_in[11];
  const float* fw2   = (const float*)d_in[12];
  const float* fb2   = (const float*)d_in[13];
  const float* lnf_s = (const float*)d_in[14];
  const float* lnf_b = (const float*)d_in[15];
  const float* wp    = (const float*)d_in[16];
  const float* bp    = (const float*)d_in[17];
  float* outp = (float*)d_out;

  char* ws = (char*)d_ws;
  size_t off = 0;
  auto alloc = [&](size_t bytes) -> void* {
    void* p = ws + off;
    off = (off + bytes + 255) & ~(size_t)255;
    return p;
  };

  const size_t nW = (size_t)kL * kD * kD;                 // per stacked tensor
  __bf16* wq_p  = (__bf16*)alloc(nW * 2);
  __bf16* wk_p  = (__bf16*)alloc(nW * 2);
  __bf16* wv_p  = (__bf16*)alloc(nW * 2);
  __bf16* fw1_p = (__bf16*)alloc(nW * 2);
  __bf16* fw2_p = (__bf16*)alloc(nW * 2);
  __bf16* wp_p  = (__bf16*)alloc((size_t)kD * kV * 2);
  float*  xbuf  = (float*)alloc((size_t)kM * kD * 4);
  __bf16* hbuf  = (__bf16*)alloc((size_t)kM * kD * 2);
  __bf16* qbuf  = (__bf16*)alloc((size_t)kM * kD * 2);
  __bf16* kbuf  = (__bf16*)alloc((size_t)kM * kD * 2);
  __bf16* vtbuf = (__bf16*)alloc((size_t)kM * kD * 2);
  __bf16* midb  = (__bf16*)alloc((size_t)kM * kD * 2);

  // 1) Repack all weights to bf16 WMMA fragment images (read-once f32).
  {
    unsigned gb = (unsigned)((nW + 255) / 256);
    pack_w_kernel<<<gb, 256, 0, stream>>>(wq,  wq_p,  kD, kD, kL);
    pack_w_kernel<<<gb, 256, 0, stream>>>(wk,  wk_p,  kD, kD, kL);
    pack_w_kernel<<<gb, 256, 0, stream>>>(wv,  wv_p,  kD, kD, kL);
    pack_w_kernel<<<gb, 256, 0, stream>>>(fw1, fw1_p, kD, kD, kL);
    pack_w_kernel<<<gb, 256, 0, stream>>>(fw2, fw2_p, kD, kD, kL);
    unsigned gp = (unsigned)(((size_t)kD * kV + 255) / 256);
    pack_w_kernel<<<gp, 256, 0, stream>>>(wp, wp_p, kD, kV, 1);
  }

  // 2) Embedding
  embed_kernel<<<kM, 256, 0, stream>>>(ctx, tok, pos, xbuf);

  const dim3 gD(kD / 128, kM / 64);   // D-wide GEMMs: 8 x 32 blocks
  const dim3 gA(kT / 64, kH, kB);     // attention

  // 3) Transformer blocks
  for (int l = 0; l < kL; ++l) {
    const size_t wo = (size_t)l * kD * kD;
    ln_kernel<<<kM, 256, 0, stream>>>(xbuf, ln1_s + l * kD, ln1_b + l * kD,
                                      hbuf);
    gemm_bf16<1, false, false, false><<<gD, 256, 0, stream>>>(
        hbuf, wq_p + wo, nullptr, nullptr, qbuf, kM, kD, kD);
    gemm_bf16<1, false, false, false><<<gD, 256, 0, stream>>>(
        hbuf, wk_p + wo, nullptr, nullptr, kbuf, kM, kD, kD);
    gemm_bf16<2, false, false, false><<<gD, 256, 0, stream>>>(
        hbuf, wv_p + wo, nullptr, nullptr, vtbuf, kM, kD, kD);
    attn_kernel<<<gA, 128, 0, stream>>>(qbuf, kbuf, vtbuf, xbuf);
    ln_kernel<<<kM, 256, 0, stream>>>(xbuf, ln2_s + l * kD, ln2_b + l * kD,
                                      hbuf);
    gemm_bf16<1, true, true, false><<<gD, 256, 0, stream>>>(
        hbuf, fw1_p + wo, fb1 + l * kD, nullptr, midb, kM, kD, kD);
    gemm_bf16<0, true, false, true><<<gD, 256, 0, stream>>>(
        midb, fw2_p + wo, fb2 + l * kD, xbuf, xbuf, kM, kD, kD);
  }

  // 4) Final LN + logits straight into d_out, softmax in place.
  ln_kernel<<<kM, 256, 0, stream>>>(xbuf, lnf_s, lnf_b, hbuf);
  const dim3 gL(kV / 128, kM / 64);   // 250 x 32 blocks
  gemm_bf16<0, true, false, false><<<gL, 256, 0, stream>>>(
      hbuf, wp_p, bp, nullptr, outp, kM, kV, kD);
  softmax_kernel<<<kM, 256, 0, stream>>>(outp, kV);
}
